// TKSA_Attention_44890998177859
// MI455X (gfx1250) — matrix operand order repaired and checked by reading, not compiled, not run on hardware
//
#include <hip/hip_runtime.h>
#include <hip/hip_bf16.h>

// ---------------------------------------------------------------------------
// CDNA5 / gfx1250 implementation of TKSA attention.
// GEMM stages: v_wmma_f32_16x16x32_bf16 (wave32, f32 accumulate).
// A-tile / q,k-tile staging: Tensor Data Mover (tensor_load_to_lds) with
// hardware LDS padding, fenced by s_wait_tensorcnt.
// ---------------------------------------------------------------------------

typedef __bf16 bf16;
typedef __attribute__((ext_vector_type(16))) __bf16 v16bf;
typedef __attribute__((ext_vector_type(8)))  float  v8f;

union FragBF16 { v16bf v; float4 f[2]; };

#define BM 64
#define BN 128
#define BK 32
#define LDS_STRIDE 40   // 32 K-elems + 8 bf16 pad (row start stays 16B aligned)

#if defined(__has_builtin)
#if __has_builtin(__builtin_amdgcn_tensor_load_to_lds) && \
    __has_builtin(__builtin_amdgcn_s_wait_tensorcnt)
#define HAVE_TDM 1
#endif
#endif

#ifdef HAVE_TDM
typedef unsigned int u32x4 __attribute__((ext_vector_type(4)));
typedef int          i32x4 __attribute__((ext_vector_type(4)));
typedef int          i32x8 __attribute__((ext_vector_type(8)));

// DMA a (tile_h x 32) bf16 tile (row stride = stride_elems) from global into
// LDS, inserting 16B of padding after every 64B row so LDS row stride is
// LDS_STRIDE(=40) bf16.  D# fields per CDNA5 ISA.
__device__ __forceinline__ void tdm_load_tile_bf16(const bf16* gptr,
                                                   unsigned lds_byte_addr,
                                                   int tile_h, int stride_elems) {
    unsigned long long ga = (unsigned long long)(uintptr_t)gptr;
    u32x4 g0;
    g0[0] = 1u;                                     // count=1, load, no gather
    g0[1] = lds_byte_addr;                          // lds_addr
    g0[2] = (unsigned)(ga & 0xFFFFFFFFu);           // global_addr[31:0]
    g0[3] = (unsigned)((ga >> 32) & 0x01FFFFFFu)    // global_addr[56:32]
          | (2u << 30);                             // type = 2 ("image")
    i32x8 g1;
    g1[0] = (int)((1u << 16)                        // data_size = 2 bytes
                | (1u << 20)                        // pad_enable
                | (3u << 22)                        // pad_interval: 16 DWORDs (64B)
                | (3u << 25));                      // pad_amount:   4 DWORDs (16B)
    g1[1] = (int)(((unsigned)stride_elems & 0xFFFFu) << 16);   // tensor_dim0 lo16
    g1[2] = (int)(((unsigned)stride_elems >> 16)               // tensor_dim0 hi16
                | (((unsigned)tile_h & 0xFFFFu) << 16));       // tensor_dim1 lo16
    g1[3] = (int)(32u << 16);                       // tensor_dim1 hi=0 | tile_dim0=32
    g1[4] = (int)((unsigned)tile_h & 0xFFFFu);      // tile_dim1 | tile_dim2=0
    g1[5] = stride_elems;                           // tensor_dim0_stride lo32
    g1[6] = 0;                                      // stride0 hi | stride1 lo
    g1[7] = 0;
    i32x4 z4 = {0, 0, 0, 0};                        // 2-D tensor: groups 2/3 unused
    i32x8 z8 = {0, 0, 0, 0, 0, 0, 0, 0};
    __builtin_amdgcn_tensor_load_to_lds(g0, g1, z4, z4, z8, 0);
}
#endif

// ---------------------------------------------------------------------------
// f32 -> bf16 convert
// ---------------------------------------------------------------------------
__global__ __launch_bounds__(256) void cvt_f32_to_bf16(const float* __restrict__ in,
                                                       bf16* __restrict__ out,
                                                       long long n) {
    long long i = (long long)blockIdx.x * 256 + threadIdx.x;
    if (i < n) out[i] = (bf16)in[i];
}

// ---------------------------------------------------------------------------
// Batched bf16 GEMM: C[z] = A[z] (MxK, row major) * B[z] (KxN, row major)
// 64x128 block tile, 8 waves in 2(M) x 4(N), each wave 32x32 = 2x2 WMMA
// tiles.  A tile staged by the TDM (async DMA, TENSORcnt), B tile staged
// with a transpose through LDS by the waves.
// ---------------------------------------------------------------------------
template <bool OUT_BF16>
__global__ __launch_bounds__(256) void gemm_bf16_wmma(
    const bf16* __restrict__ A, int lda, long long azs,
    const bf16* __restrict__ B, int ldb, long long bzs,
    void* __restrict__ Cv, int ldc, long long czs,
    int M, int N, int K)
{
    __shared__ bf16 lA[BM * LDS_STRIDE];
    __shared__ bf16 lB[BN * LDS_STRIDE];

    const int tid  = threadIdx.x;
    const int lane = tid & 31;
    const int wv   = tid >> 5;     // 0..7
    const int wm   = wv & 1;       // wave row (M)
    const int wn   = wv >> 1;      // wave col (N)
    const int l15  = lane & 15;
    const int half = lane >> 4;

    const int z  = blockIdx.z;
    const int m0 = blockIdx.y * BM;
    const int n0 = blockIdx.x * BN;

    A += (long long)z * azs;
    B += (long long)z * bzs;

    // B-tile loaders: 32 rows(k) x 128 cols(n), 8 threads/row, 16 bf16 each
    const int br = tid >> 3;
    const int bc = (tid & 7) * 16;
    const bf16* srcB = B + (long long)br * ldb + (n0 + bc);

    const bf16* srcA0 = A + (long long)m0 * lda;      // tile row 0
#ifdef HAVE_TDM
    const unsigned lAoff = (unsigned)(uintptr_t)&lA[0];  // flat->LDS aperture: low 32b
#else
    // A-tile loaders: 64 rows x 32 cols, 4 threads/row, 8 bf16 (16B) each
    const int ar = tid >> 2;
    const int ac = (tid & 3) * 8;
    const bf16* srcA = srcA0 + (long long)ar * lda + ac;
#endif

    v8f acc[2][2] = {};

    for (int k0 = 0; k0 < K; k0 += BK) {
        // ---- stage A tile (row major [m][k], padded rows) ----
#ifdef HAVE_TDM
        if (tid == 0)
            tdm_load_tile_bf16(srcA0 + k0, lAoff, BM, lda);
#else
        *(float4*)&lA[ar * LDS_STRIDE + ac] = *(const float4*)(srcA + k0);
#endif
        // ---- stage B tile transposed into [n][k] ----
        {
            const bf16* p = srcB + (long long)k0 * ldb;
            float4 v0 = *(const float4*)p;
            float4 v1 = *(const float4*)(p + 8);
            const bf16* e0 = (const bf16*)&v0;
            const bf16* e1 = (const bf16*)&v1;
#pragma unroll
            for (int i = 0; i < 8; ++i) lB[(bc + i) * LDS_STRIDE + br] = e0[i];
#pragma unroll
            for (int i = 0; i < 8; ++i) lB[(bc + 8 + i) * LDS_STRIDE + br] = e1[i];
        }
        if (k0 + BK < K)
            __builtin_prefetch(srcB + (long long)(k0 + BK) * ldb, 0, 0); // global_prefetch_b8
#ifdef HAVE_TDM
        __builtin_amdgcn_s_wait_tensorcnt(0);
#endif
        __syncthreads();

        // ---- fragments per ISA VGPR layouts ----
        FragBF16 af[2], bfg[2];
#pragma unroll
        for (int i = 0; i < 2; ++i) {
            // A 16x32: lanes 0-15 -> K{0..7,16..23}; lanes 16-31 -> K{8..15,24..31}
            const float4* p = (const float4*)&lA[(wm * 32 + i * 16 + l15) * LDS_STRIDE];
            af[i].f[0] = p[half];
            af[i].f[1] = p[2 + half];
        }
#pragma unroll
        for (int j = 0; j < 2; ++j) {
            // B 32x16: lanes 0-15 -> K{0..7,16..23}; lanes 16-31 -> K{8..15,24..31} (col = l15)
            const float4* p = (const float4*)&lB[(wn * 32 + j * 16 + l15) * LDS_STRIDE];
            bfg[j].f[0] = p[half];
            bfg[j].f[1] = p[2 + half];
        }
#pragma unroll
        for (int i = 0; i < 2; ++i)
#pragma unroll
            for (int j = 0; j < 2; ++j)
                acc[i][j] = __builtin_amdgcn_wmma_f32_16x16x32_bf16(
                    false, af[i].v, false, bfg[j].v, (short)0, acc[i][j], false, false);
        __syncthreads();
    }

    // ---- epilogue: C VGPR r, lanes 0-15 -> M=r, lanes 16-31 -> M=r+8 ----
#pragma unroll
    for (int i = 0; i < 2; ++i) {
#pragma unroll
        for (int j = 0; j < 2; ++j) {
            const int mbase = m0 + wm * 32 + i * 16 + 8 * half;
            const int nc    = n0 + wn * 32 + j * 16 + l15;
#pragma unroll
            for (int r = 0; r < 8; ++r) {
                const long long off = (long long)(mbase + r) * ldc + nc;
                if (OUT_BF16) {
                    bf16* C = (bf16*)Cv + (long long)z * czs;
                    C[off] = (bf16)acc[i][j][r];
                } else {
                    float* C = (float*)Cv + (long long)z * czs;
                    C[off] = acc[i][j][r];
                }
            }
        }
    }
}

// ---------------------------------------------------------------------------
// 3x3 depthwise conv ('SAME', zero pad) over (b,1536,64,64) bf16; output is
// remapped to (z=b*8+head, which{q,k,v}, cph, n) so q/k/v rows are contiguous
// along n for the attention GEMMs.
// ---------------------------------------------------------------------------
__global__ __launch_bounds__(256) void dwconv3x3(const bf16* __restrict__ qkv,
                                                 const float* __restrict__ w,
                                                 bf16* __restrict__ out) {
    long long idx = (long long)blockIdx.x * 256 + threadIdx.x;   // 8*1536*4096
    int n = (int)(idx & 4095);
    long long ch = idx >> 12;
    int c = (int)(ch % 1536);
    int b = (int)(ch / 1536);
    int h = n >> 6, x = n & 63;

    const bf16*  in = qkv + ((long long)b * 1536 + c) * 4096;
    const float* wc = w + c * 9;
    float s = 0.f;
#pragma unroll
    for (int dy = -1; dy <= 1; ++dy) {
#pragma unroll
        for (int dx = -1; dx <= 1; ++dx) {
            int hh = h + dy, ww = x + dx;
            if (hh >= 0 && hh < 64 && ww >= 0 && ww < 64)
                s += wc[(dy + 1) * 3 + (dx + 1)] * (float)in[hh * 64 + ww];
        }
    }
    int which = c >> 9;
    int cr = c & 511;
    int head = cr >> 6;
    int cph = cr & 63;
    long long z = (long long)b * 8 + head;
    out[((z * 3 + which) * 64 + cph) * 4096 + n] = (bf16)s;
}

// ---------------------------------------------------------------------------
// Row inverse L2 norms of q and k (64 z * {q,k} * 64 rows, K=4096)
// ---------------------------------------------------------------------------
__global__ __launch_bounds__(256) void rownorm(const bf16* __restrict__ qkv_dw,
                                               float* __restrict__ invq,
                                               float* __restrict__ invk) {
    int r = blockIdx.x;            // 0..8191
    int z = r >> 7;
    int rem = r & 127;
    int which = rem >> 6;          // 0=q, 1=k
    int c = rem & 63;
    const bf16* row = qkv_dw + (((long long)z * 3 + which) * 64 + c) * 4096;

    float ss = 0.f;
    for (int i = threadIdx.x; i < 4096; i += 256) {
        float v = (float)row[i];
        ss += v * v;
    }
    __shared__ float red[256];
    red[threadIdx.x] = ss;
    __syncthreads();
    for (int s = 128; s > 0; s >>= 1) {
        if (threadIdx.x < (unsigned)s) red[threadIdx.x] += red[threadIdx.x + s];
        __syncthreads();
    }
    if (threadIdx.x == 0) {
        float inv = 1.0f / fmaxf(sqrtf(red[0]), 1e-12f);
        (which ? invk : invq)[z * 64 + c] = inv;
    }
}

// ---------------------------------------------------------------------------
// Fused attention: attn = (q kT) * temp * invq * invk via WMMA over K=4096
// (q,k tiles DMA'd by the TDM), then per-row top-k thresholds {32,42,48,51},
// 4 masked softmaxes combined with a1..a4 into one 64x64 bf16 matrix W per z.
// ---------------------------------------------------------------------------
__global__ __launch_bounds__(256) void attn_topk_softmax(
    const bf16* __restrict__ qkv_dw,
    const float* __restrict__ invq, const float* __restrict__ invk,
    const float* __restrict__ temperature,
    const float* __restrict__ a1, const float* __restrict__ a2,
    const float* __restrict__ a3, const float* __restrict__ a4,
    bf16* __restrict__ Wout)
{
    __shared__ bf16  lq[64 * LDS_STRIDE];
    __shared__ bf16  lk[64 * LDS_STRIDE];
    __shared__ float sattn[64 * 65];

    const int z    = blockIdx.x;
    const int head = z & 7;
    const int tid  = threadIdx.x;
    const int lane = tid & 31;
    const int wv   = tid >> 5;
    const int l15  = lane & 15;
    const int half = lane >> 4;
    const int tm   = wv & 3;            // 4 M-tiles
    const int tnb  = (wv >> 2) * 2;     // 2 N-tiles per wave

    const bf16* qb = qkv_dw + ((long long)z * 3 + 0) * 64 * 4096;
    const bf16* kb = qkv_dw + ((long long)z * 3 + 1) * 64 * 4096;

#ifdef HAVE_TDM
    const unsigned lqOff = (unsigned)(uintptr_t)&lq[0];
    const unsigned lkOff = (unsigned)(uintptr_t)&lk[0];
#else
    const int lr = tid >> 2;
    const int lc = (tid & 3) * 8;
#endif

    v8f acc[2] = {};

    for (int k0 = 0; k0 < 4096; k0 += BK) {
#ifdef HAVE_TDM
        if (tid == 0) {
            tdm_load_tile_bf16(qb + k0, lqOff, 64, 4096);
            tdm_load_tile_bf16(kb + k0, lkOff, 64, 4096);
        }
        __builtin_amdgcn_s_wait_tensorcnt(0);
#else
        *(float4*)&lq[lr * LDS_STRIDE + lc] = *(const float4*)(qb + (long long)lr * 4096 + k0 + lc);
        *(float4*)&lk[lr * LDS_STRIDE + lc] = *(const float4*)(kb + (long long)lr * 4096 + k0 + lc);
#endif
        __syncthreads();

        FragBF16 a, b0, b1;
        const float4* pa = (const float4*)&lq[(tm * 16 + l15) * LDS_STRIDE];
        a.f[0] = pa[half];
        a.f[1] = pa[2 + half];
        // k tile is stored [d][n] which is exactly the B [ncol][k] layout
        const float4* pb0 = (const float4*)&lk[(tnb * 16 + l15) * LDS_STRIDE];
        b0.f[0] = pb0[half]; b0.f[1] = pb0[2 + half];
        const float4* pb1 = (const float4*)&lk[((tnb + 1) * 16 + l15) * LDS_STRIDE];
        b1.f[0] = pb1[half]; b1.f[1] = pb1[2 + half];

        acc[0] = __builtin_amdgcn_wmma_f32_16x16x32_bf16(false, a.v, false, b0.v, (short)0, acc[0], false, false);
        acc[1] = __builtin_amdgcn_wmma_f32_16x16x32_bf16(false, a.v, false, b1.v, (short)0, acc[1], false, false);
        __syncthreads();
    }

    // epilogue: fold temperature and inverse norms, park in LDS
    const float tmp = temperature[head];
#pragma unroll
    for (int j = 0; j < 2; ++j) {
        const int d = (tnb + j) * 16 + l15;
        const float sk = invk[z * 64 + d];
#pragma unroll
        for (int r = 0; r < 8; ++r) {
            const int c = tm * 16 + r + 8 * half;
            sattn[c * 65 + d] = acc[j][r] * tmp * invq[z * 64 + c] * sk;
        }
    }
    __syncthreads();

    // per-row top-k + 4 softmaxes (rows of 64; threads 0..63)
    if (tid < 64) {
        const float* arow = &sattn[tid * 65];
        const int kt[4] = {32, 42, 48, 51};
        float thr[4];
        unsigned long long removed = 0ull;
        float rowmax = -3.4e38f;
        int ti = 0;
        for (int it = 1; it <= 51; ++it) {
            float m = -3.4e38f; int mi = 0;
            for (int j = 0; j < 64; ++j) {
                if (!((removed >> j) & 1ull)) {
                    float v = arow[j];
                    if (v > m) { m = v; mi = j; }
                }
            }
            removed |= 1ull << mi;
            if (it == 1) rowmax = m;
            if (ti < 4 && it == kt[ti]) { thr[ti] = m; ++ti; }
        }
        float Z[4] = {0.f, 0.f, 0.f, 0.f};
        for (int j = 0; j < 64; ++j) {
            float v = arow[j];
            float e = __expf(v - rowmax);
#pragma unroll
            for (int i = 0; i < 4; ++i)
                if (v >= thr[i]) Z[i] += e;
        }
        const float av[4] = {a1[0], a2[0], a3[0], a4[0]};
        float wz[4];
#pragma unroll
        for (int i = 0; i < 4; ++i) wz[i] = av[i] / Z[i];
        bf16* wrow = Wout + (long long)z * 4096 + tid * 64;
        for (int j = 0; j < 64; ++j) {
            float v = arow[j];
            float e = __expf(v - rowmax);
            float s = 0.f;
#pragma unroll
            for (int i = 0; i < 4; ++i)
                if (v >= thr[i]) s += wz[i] * e;
            wrow[j] = (bf16)s;
        }
    }
}

// ---------------------------------------------------------------------------
// Launcher
// ---------------------------------------------------------------------------
extern "C" void kernel_launch(void* const* d_in, const int* in_sizes, int n_in,
                              void* d_out, int out_size, void* d_ws, size_t ws_size,
                              hipStream_t stream) {
    (void)in_sizes; (void)n_in; (void)out_size; (void)ws_size;

    const float* x      = (const float*)d_in[0];   // (8,512,64,64)
    const float* qkv_w  = (const float*)d_in[1];   // (1536,512,1,1)
    const float* dw_w   = (const float*)d_in[2];   // (1536,1,3,3)
    const float* proj_w = (const float*)d_in[3];   // (512,512,1,1)
    const float* temp   = (const float*)d_in[4];   // (8,1,1)
    const float* a1     = (const float*)d_in[5];
    const float* a2     = (const float*)d_in[6];
    const float* a3     = (const float*)d_in[7];
    const float* a4     = (const float*)d_in[8];

    const long long NX   = 8ll * 512 * 4096;       // x / out_attn elems
    const long long NQKV = 8ll * 1536 * 4096;      // qkv elems

    // workspace layout (bytes, 256-aligned)
    char* ws = (char*)d_ws;
    bf16*  qkv_b   = (bf16*)(ws + 0);                       // 96 MB (dead after dwconv)
    bf16*  outatt  = (bf16*)(ws + 0);                       // 32 MB (reuses qkv_b region)
    bf16*  dw_b    = (bf16*)(ws + (NQKV * 2));              // 96 MB
    bf16*  x_b     = (bf16*)(ws + (NQKV * 4));              // 32 MB
    bf16*  wqkv_b  = (bf16*)(ws + (NQKV * 4) + NX * 2);                 // 1.5 MB
    bf16*  wproj_b = (bf16*)((char*)wqkv_b + 1536ll * 512 * 2);         // 0.5 MB
    float* invq    = (float*)((char*)wproj_b + 512ll * 512 * 2);        // 16 KB
    float* invk    = (float*)((char*)invq + 64 * 64 * 4);               // 16 KB
    bf16*  W_b     = (bf16*)((char*)invk + 64 * 64 * 4);                // 0.5 MB

    // 1) f32 -> bf16 conversions
    cvt_f32_to_bf16<<<(unsigned)((NX + 255) / 256), 256, 0, stream>>>(x, x_b, NX);
    cvt_f32_to_bf16<<<(1536 * 512 + 255) / 256, 256, 0, stream>>>(qkv_w, wqkv_b, 1536ll * 512);
    cvt_f32_to_bf16<<<(512 * 512 + 255) / 256, 256, 0, stream>>>(proj_w, wproj_b, 512ll * 512);

    // 2) qkv 1x1 conv: (1536x512) x (512x4096) per batch, out bf16
    gemm_bf16_wmma<true><<<dim3(32, 24, 8), 256, 0, stream>>>(
        wqkv_b, 512, 0ll,
        x_b, 4096, 512ll * 4096,
        qkv_b, 4096, 1536ll * 4096,
        1536, 4096, 512);

    // 3) 3x3 depthwise + layout remap to (z, q/k/v, cph, n)
    dwconv3x3<<<(unsigned)(NQKV / 256), 256, 0, stream>>>(qkv_b, dw_w, dw_b);

    // 4) inverse row norms of q, k
    rownorm<<<8192, 256, 0, stream>>>(dw_b, invq, invk);

    // 5) attn + top-k masked softmaxes -> combined W (bf16, 64x64 per z)
    attn_topk_softmax<<<64, 256, 0, stream>>>(dw_b, invq, invk, temp,
                                              a1, a2, a3, a4, W_b);

    // 6) out = W @ v  : (64x64) x (64x4096) per z, out bf16 into (b,c,n) layout
    gemm_bf16_wmma<true><<<dim3(32, 1, 64), 256, 0, stream>>>(
        W_b, 64, 64ll * 64,
        dw_b + 2ll * 64 * 4096, 4096, 3ll * 64 * 4096,
        outatt, 4096, 64ll * 4096,
        64, 4096, 64);

    // 7) proj 1x1 conv: (512x512) x (512x4096) per batch, out f32 -> d_out
    gemm_bf16_wmma<false><<<dim3(32, 8, 8), 256, 0, stream>>>(
        wproj_b, 512, 0ll,
        outatt, 4096, 512ll * 4096,
        d_out, 4096, 512ll * 4096,
        512, 4096, 512);
}
